// NetVLADPool_53979148976680
// MI455X (gfx1250) — matrix operand order repaired and checked by reading, not compiled, not run on hardware
//
#include <hip/hip_runtime.h>

typedef __attribute__((ext_vector_type(16))) _Float16 v16h;
typedef __attribute__((ext_vector_type(8)))  _Float16 v8h;
typedef __attribute__((ext_vector_type(8)))  float    v8f;
typedef __attribute__((ext_vector_type(4)))  float    v4f;

#define N_PTS  2048
#define CDIM   64
#define KDIM   32
#define NT     128
#define NTILES (N_PTS / NT)

// Build a 16x16x32 f16 A/B operand from two 16-byte LDS chunks.
// A-operand pattern: elements 0..7 at p[0..7], elements 8..15 at p[16..23].
__device__ __forceinline__ v16h make_gap16(const _Float16* p) {
    v8h lo = *(const v8h*)p;
    v8h hi = *(const v8h*)(p + 16);
    return __builtin_shufflevector(lo, hi, 0,1,2,3,4,5,6,7,8,9,10,11,12,13,14,15);
}
// Contiguous 16 halves (32 bytes, only 16B-alignment guaranteed).
__device__ __forceinline__ v16h make_seq16(const _Float16* p) {
    v8h lo = *(const v8h*)p;
    v8h hi = *(const v8h*)(p + 8);
    return __builtin_shufflevector(lo, hi, 0,1,2,3,4,5,6,7,8,9,10,11,12,13,14,15);
}

__global__ __launch_bounds__(256)
void netvlad_fused_kernel(const float* __restrict__ R,
                          const float* __restrict__ W,
                          const float* __restrict__ bias,
                          const float* __restrict__ cent,
                          float* __restrict__ out)
{
    // r tile row-major (GEMM1 A operand): stride 72 halves = 144B (16B aligned rows)
    __shared__ __attribute__((aligned(16))) _Float16 sR [NT][72];
    // r tile transposed [c][n] (GEMM2 B operand): stride 136 halves = 272B
    __shared__ __attribute__((aligned(16))) _Float16 sRT[CDIM][136];
    // softmax probs transposed [k][n] (GEMM2 A operand)
    __shared__ __attribute__((aligned(16))) _Float16 sA [KDIM][136];
    // logits (fp32) for softmax
    __shared__ float sLog[NT][KDIM + 1];
    // W pre-swizzled into per-(c-chunk kk, k-tile ki, lane) contiguous B layout
    __shared__ __attribute__((aligned(16))) _Float16 sWB[2][2][32][16];
    __shared__ float sB[KDIM];
    __shared__ float sSum[KDIM];

    const int tid  = threadIdx.x;
    const int lane = tid & 31;
    const int wave = tid >> 5;
    const int m    = blockIdx.x;
    const int hv   = (lane >> 4) & 1;   // which half-wave (ISA lane-split)
    const int l15  = lane & 15;

    // ---- Preload W into B-ready layout; bias; zero asum ----
    // B k order used here (contiguous, same as A; not the ISA table order): lanes 0-15 hold K=0..15, lanes 16-31 hold K=16..31,
    // element e -> K = 16*halfwave + e. Here the B "K" dim is c within a 32-chunk.
    #pragma unroll
    for (int i = 0; i < 8; ++i) {
        int flat = tid * 8 + i;               // 0..2047
        int e    = flat & 15;
        int ln   = (flat >> 4) & 31;
        int ki   = (flat >> 9) & 1;
        int kk   = (flat >> 10) & 1;
        int c    = kk * 32 + ((ln >> 4) & 1) * 16 + e;
        int kcol = ki * 16 + (ln & 15);
        sWB[kk][ki][ln][e] = (_Float16)W[kcol * CDIM + c];
    }
    if (tid < KDIM) { sB[tid] = bias[tid]; sSum[tid] = 0.0f; }

    // GEMM2 persistent accumulator: wave -> v tile (mi2 = K-row tile, ci = C-col tile)
    v8f vacc = {};
    const int mi2 = wave >> 2;
    const int ci  = wave & 3;

    const float* rm = R + (size_t)m * N_PTS * CDIM;

    for (int t = 0; t < NTILES; ++t) {
        const float* rb = rm + t * NT * CDIM;

        // ---- Stream r tile: global fp32 -> LDS f16, row-major + transposed ----
        #pragma unroll
        for (int jj = 0; jj < 8; ++jj) {
            int i4  = tid + jj * 256;         // float4 index, 0..2047
            int row = i4 >> 4;
            int c4  = (i4 & 15) * 4;
            v4f d = *(const v4f*)(rb + row * CDIM + c4);
            #pragma unroll
            for (int q = 0; q < 4; ++q) {
                _Float16 h = (_Float16)d[q];
                sR [row][c4 + q]  = h;
                sRT[c4 + q][row]  = h;
            }
        }
        if (t + 1 < NTILES)
            __builtin_prefetch(rb + NT * CDIM + tid * 32, 0, 1);  // global_prefetch_b8
        __syncthreads();

        // ---- GEMM1: logits(128x32) = r(128x64) @ W^T(64x32), f16 WMMA ----
        // 16 output tiles (8 M x 2 Kcol); this wave does ot = wave, wave+8.
        #pragma unroll
        for (int oo = 0; oo < 2; ++oo) {
            int ot = wave + oo * 8;
            int mi = ot >> 1;
            int ki = ot & 1;
            v8f acc = {};
            #pragma unroll
            for (int kk = 0; kk < 2; ++kk) {
                v16h Av = make_seq16(&sR[mi * 16 + l15][kk * 32 + hv * 16]);
                v16h Bv = make_seq16(&sWB[kk][ki][lane][0]);
                acc = __builtin_amdgcn_wmma_f32_16x16x32_f16(
                          false, Av, false, Bv, (short)0, acc, false, false);
            }
            // D layout: VGPR j -> M = j + 8*halfwave, N = lane&15
            #pragma unroll
            for (int j = 0; j < 8; ++j)
                sLog[mi * 16 + j + hv * 8][ki * 16 + l15] = acc[j];
        }
        __syncthreads();

        // ---- Softmax over K per row (ALPHA == 1), store a^T as f16 ----
        if (tid < NT) {
            float vals[KDIM];
            float mx = -3.4e38f;
            #pragma unroll
            for (int k = 0; k < KDIM; ++k) {
                vals[k] = sLog[tid][k] + sB[k];
                mx = fmaxf(mx, vals[k]);
            }
            float s = 0.0f;
            #pragma unroll
            for (int k = 0; k < KDIM; ++k) { vals[k] = __expf(vals[k] - mx); s += vals[k]; }
            float inv = __builtin_amdgcn_rcpf(s);
            #pragma unroll
            for (int k = 0; k < KDIM; ++k)
                sA[k][tid] = (_Float16)(vals[k] * inv);
        }
        __syncthreads();

        // ---- asum[k] += sum_n a[n,k] (per-thread partials + ds_add_f32) ----
        {
            int k   = tid >> 3;
            int seg = (tid & 7) * 16;
            float part = 0.0f;
            #pragma unroll
            for (int q = 0; q < 16; ++q) part += (float)sA[k][seg + q];
            atomicAdd(&sSum[k], part);
        }

        // ---- GEMM2: v(32x64) += a^T(32x128) @ r(128x64), 1 tile per wave ----
        #pragma unroll
        for (int nn = 0; nn < NT; nn += 32) {
            v16h Av = make_seq16(&sA [mi2 * 16 + l15][nn + hv * 16]);
            v16h Bv = make_seq16(&sRT[ci  * 16 + l15][nn + hv * 16]);
            vacc = __builtin_amdgcn_wmma_f32_16x16x32_f16(
                       false, Av, false, Bv, (short)0, vacc, false, false);
        }
        __syncthreads();   // protect sR/sRT/sA/sLog for next tile; publishes sSum
    }

    // ---- Finalize: v[k][c] -= asum[k] * centroids[k][c]; write out ----
    float* om = out + (size_t)m * KDIM * CDIM;
    #pragma unroll
    for (int j = 0; j < 8; ++j) {
        int k = mi2 * 16 + j + hv * 8;
        int c = ci * 16 + l15;
        om[k * CDIM + c] = vacc[j] - sSum[k] * cent[k * CDIM + c];
    }
}

extern "C" void kernel_launch(void* const* d_in, const int* in_sizes, int n_in,
                              void* d_out, int out_size, void* d_ws, size_t ws_size,
                              hipStream_t stream) {
    const float* R    = (const float*)d_in[0];   // (8,32,2048,64) fp32
    const float* W    = (const float*)d_in[1];   // (32,64)
    const float* b    = (const float*)d_in[2];   // (32,)
    const float* cent = (const float*)d_in[3];   // (32,64)
    float* out = (float*)d_out;                  // (8,32,32,64)

    netvlad_fused_kernel<<<dim3(8 * 32), dim3(256), 0, stream>>>(R, W, b, cent, out);
}
